// Head_89661737271479
// MI455X (gfx1250) — compile-verified, matrix operand order checked by reading, not run on hardware
//
#include <hip/hip_runtime.h>
#include <hip/hip_bf16.h>

typedef __attribute__((ext_vector_type(16))) _Float16 v16h;
typedef __attribute__((ext_vector_type(8)))  _Float16 v8h;
typedef __attribute__((ext_vector_type(8)))  float    v8f;

#define TSEQ 2048
#define NB   16
#define DDIM 768
#define HSD  96

__device__ __forceinline__ v8f wmma_f16(v16h a, v16h b, v8f c) {
    // (neg_a, A, neg_b, B, c_mod, C, reuse_a, reuse_b)
    return __builtin_amdgcn_wmma_f32_16x16x32_f16(false, a, false, b, (short)0, c, false, false);
}

// ---------------------------------------------------------------------------
// Projection: Y = X * W for W in {Wq, Wk, Wv}.  X:[32768,768] f32, W:[768,96] f32.
// One wave per (mat, mtile, ntile). Writes f16: Qh/Kh row-major [M][96],
// Vt transposed per batch [b][96][2048].
// ---------------------------------------------------------------------------
__global__ __launch_bounds__(256) void proj_kernel(
    const float* __restrict__ x,
    const float* __restrict__ Wq, const float* __restrict__ Wk, const float* __restrict__ Wv,
    _Float16* __restrict__ Qh, _Float16* __restrict__ Kh, _Float16* __restrict__ Vt)
{
    const int lane  = threadIdx.x & 31;
    const int wib   = threadIdx.x >> 5;
    const int wave  = blockIdx.x * 8 + wib;

    const int NT = HSD / 16;            // 6
    const int MT = (NB * TSEQ) / 16;    // 2048
    int tmp = wave;
    const int nt  = tmp % NT; tmp /= NT;
    const int mt  = tmp % MT; tmp /= MT;
    const int mat = tmp;                // 0=Q 1=K 2=V
    if (mat >= 3) return;

    const float* W = (mat == 0) ? Wq : (mat == 1) ? Wk : Wv;

    const int half  = lane >> 4;        // 0/1 selects K-subgroup
    const int l16   = lane & 15;
    const int kbase = half * 8;
    const int mrow  = mt * 16 + l16;    // A-matrix row held by this lane
    const int ncol  = nt * 16 + l16;    // B/D column held by this lane

    v8f acc;
    #pragma unroll
    for (int r = 0; r < 8; ++r) acc[r] = 0.f;

    for (int kk = 0; kk < DDIM / 32; ++kk) {          // 24 iterations
        const int d0 = kk * 32 + kbase;
        // A fragment: elements 0..7 -> d0+0..7 ; 8..15 -> d0+16..23 (contiguous f32)
        v8f alo = *(const v8f*)(x + (size_t)mrow * DDIM + d0);
        v8f ahi = *(const v8f*)(x + (size_t)mrow * DDIM + d0 + 16);
        v16h a;
        #pragma unroll
        for (int i = 0; i < 8; ++i) { a[i] = (_Float16)alo[i]; a[8 + i] = (_Float16)ahi[i]; }
        // B fragment: contraction index d strided by 96 floats (W is tiny -> L2 resident)
        v16h bf;
        #pragma unroll
        for (int i = 0; i < 8; ++i) {
            bf[i]     = (_Float16)W[(size_t)(d0 + i)      * HSD + ncol];
            bf[8 + i] = (_Float16)W[(size_t)(d0 + 16 + i) * HSD + ncol];
        }
        acc = wmma_f16(a, bf, acc);
    }

    if (mat == 2) {
        // V transposed per batch: Vt[b][hs][t]
        #pragma unroll
        for (int r = 0; r < 8; ++r) {
            int M = mt * 16 + r + 8 * half;
            int b = M >> 11;            // /2048
            int t = M & (TSEQ - 1);
            Vt[((size_t)b * HSD + ncol) * TSEQ + t] = (_Float16)acc[r];
        }
    } else {
        _Float16* Y = (mat == 0) ? Qh : Kh;
        #pragma unroll
        for (int r = 0; r < 8; ++r) {
            int M = mt * 16 + r + 8 * half;
            Y[(size_t)M * HSD + ncol] = (_Float16)acc[r];
        }
    }
}

// ---------------------------------------------------------------------------
// Flash attention: one wave per (batch, 16-query tile), 32-key tiles.
// ---------------------------------------------------------------------------
#define ATT_WAVES 4

__global__ __launch_bounds__(32 * ATT_WAVES) void attn_kernel(
    const _Float16* __restrict__ Qh, const _Float16* __restrict__ Kh,
    const _Float16* __restrict__ Vt, float* __restrict__ out)
{
    __shared__ _Float16 pshare[ATT_WAVES][16][32] __attribute__((aligned(16)));

    const int lane  = threadIdx.x & 31;
    const int w     = threadIdx.x >> 5;
    const int wave  = blockIdx.x * ATT_WAVES + w;
    const int qt    = wave & (TSEQ / 16 - 1);   // 0..127
    const int b     = wave >> 7;
    const int l16   = lane & 15;
    const int half  = lane >> 4;
    const int kbase = half * 8;
    const int qbase = qt * 16;
    const float scale   = 0.10206207261596577f; // 96^-0.5
    const float NEG_INF = -__builtin_inff();

    const _Float16* Qb = Qh + (size_t)b * TSEQ * HSD;
    const _Float16* Kb = Kh + (size_t)b * TSEQ * HSD;
    const _Float16* Vb = Vt + (size_t)b * HSD * TSEQ;

    // Q A-fragments (row-major f16, contiguous 16B per half-fragment)
    v16h qa[3];
    {
        const int m = qbase + l16;
        #pragma unroll
        for (int c = 0; c < 3; ++c) {
            v8h lo = *(const v8h*)(Qb + (size_t)m * HSD + c * 32 + kbase);
            v8h hi = *(const v8h*)(Qb + (size_t)m * HSD + c * 32 + kbase + 16);
            #pragma unroll
            for (int i = 0; i < 8; ++i) { qa[c][i] = lo[i]; qa[c][8 + i] = hi[i]; }
        }
    }

    v8f o[6];
    #pragma unroll
    for (int h = 0; h < 6; ++h)
        #pragma unroll
        for (int r = 0; r < 8; ++r) o[h][r] = 0.f;
    float mrow[8], lrow[8];
    #pragma unroll
    for (int r = 0; r < 8; ++r) { mrow[r] = NEG_INF; lrow[r] = 0.f; }

    const int ntiles = (qbase + 15) / 32 + 1;   // causal bound on 32-key tiles
    for (int j = 0; j < ntiles; ++j) {
        const int kstart = j * 32;

        // ---- S = Q * K^T : 16x32 as two 16x16 tiles ----
        v8f s0, s1;
        #pragma unroll
        for (int r = 0; r < 8; ++r) { s0[r] = 0.f; s1[r] = 0.f; }
        #pragma unroll
        for (int c = 0; c < 3; ++c) {
            const _Float16* K0 = Kb + (size_t)(kstart + l16) * HSD + c * 32 + kbase;
            v8h lo0 = *(const v8h*)(K0);
            v8h hi0 = *(const v8h*)(K0 + 16);
            v16h b0;
            #pragma unroll
            for (int i = 0; i < 8; ++i) { b0[i] = lo0[i]; b0[8 + i] = hi0[i]; }
            s0 = wmma_f16(qa[c], b0, s0);

            const _Float16* K1 = Kb + (size_t)(kstart + 16 + l16) * HSD + c * 32 + kbase;
            v8h lo1 = *(const v8h*)(K1);
            v8h hi1 = *(const v8h*)(K1 + 16);
            v16h b1;
            #pragma unroll
            for (int i = 0; i < 8; ++i) { b1[i] = lo1[i]; b1[8 + i] = hi1[i]; }
            s1 = wmma_f16(qa[c], b1, s1);
        }

        #pragma unroll
        for (int r = 0; r < 8; ++r) { s0[r] *= scale; s1[r] *= scale; }

        if (kstart + 31 > qbase) {              // uniform branch: diagonal tiles only
            #pragma unroll
            for (int r = 0; r < 8; ++r) {
                int qglob = qbase + r + 8 * half;
                if (kstart + l16      > qglob) s0[r] = NEG_INF;
                if (kstart + 16 + l16 > qglob) s1[r] = NEG_INF;
            }
        }

        // ---- online softmax (row = r + 8*half; columns live across 16 lanes) ----
        #pragma unroll
        for (int r = 0; r < 8; ++r) {
            float mx = fmaxf(s0[r], s1[r]);
            mx = fmaxf(mx, __shfl_xor(mx, 1, 16));
            mx = fmaxf(mx, __shfl_xor(mx, 2, 16));
            mx = fmaxf(mx, __shfl_xor(mx, 4, 16));
            mx = fmaxf(mx, __shfl_xor(mx, 8, 16));
            float mnew  = fmaxf(mrow[r], mx);
            float alpha = __expf(mrow[r] - mnew);
            mrow[r] = mnew;
            float p0 = __expf(s0[r] - mnew);
            float p1 = __expf(s1[r] - mnew);
            s0[r] = p0; s1[r] = p1;
            float rs = p0 + p1;
            rs += __shfl_xor(rs, 1, 16);
            rs += __shfl_xor(rs, 2, 16);
            rs += __shfl_xor(rs, 4, 16);
            rs += __shfl_xor(rs, 8, 16);
            lrow[r] = lrow[r] * alpha + rs;
            #pragma unroll
            for (int h = 0; h < 6; ++h) o[h][r] *= alpha;
        }

        // ---- relayout P: C/D layout -> A layout via per-wave LDS bounce ----
        #pragma unroll
        for (int r = 0; r < 8; ++r) {
            int m = r + 8 * half;
            pshare[w][m][l16]      = (_Float16)s0[r];
            pshare[w][m][16 + l16] = (_Float16)s1[r];
        }
        asm volatile("s_wait_dscnt 0x0" ::: "memory");  // same-wave LDS RAW fence
        v16h pa;
        {
            v8h lo = *(const v8h*)(&pshare[w][l16][kbase]);
            v8h hi = *(const v8h*)(&pshare[w][l16][16 + kbase]);
            #pragma unroll
            for (int i = 0; i < 8; ++i) { pa[i] = lo[i]; pa[8 + i] = hi[i]; }
        }

        // ---- O += P * V (B fragments contiguous thanks to V^T layout) ----
        #pragma unroll
        for (int h = 0; h < 6; ++h) {
            const _Float16* Vp = Vb + (size_t)(h * 16 + l16) * TSEQ + kstart + kbase;
            v8h lo = *(const v8h*)(Vp);
            v8h hi = *(const v8h*)(Vp + 16);
            v16h bv;
            #pragma unroll
            for (int i = 0; i < 8; ++i) { bv[i] = lo[i]; bv[8 + i] = hi[i]; }
            o[h] = wmma_f16(pa, bv, o[h]);
        }
    }

    // ---- finalize: O / l, fp32 store ----
    float* Ob = out + (size_t)b * TSEQ * HSD;
    #pragma unroll
    for (int r = 0; r < 8; ++r) {
        float inv = 1.0f / lrow[r];
        int m = qbase + r + 8 * half;
        #pragma unroll
        for (int h = 0; h < 6; ++h)
            Ob[(size_t)m * HSD + h * 16 + l16] = o[h][r] * inv;
    }
}

// ---------------------------------------------------------------------------
extern "C" void kernel_launch(void* const* d_in, const int* in_sizes, int n_in,
                              void* d_out, int out_size, void* d_ws, size_t ws_size,
                              hipStream_t stream) {
    const float* x  = (const float*)d_in[0];
    const float* Wk = (const float*)d_in[1];
    const float* Wq = (const float*)d_in[2];
    const float* Wv = (const float*)d_in[3];
    float* out = (float*)d_out;

    // Workspace: f16 Qh | Kh (row-major [B*T][96]) | Vt ([B][96][T]) = ~18.9 MB
    const size_t per = (size_t)NB * TSEQ * HSD;
    _Float16* Qh = (_Float16*)d_ws;
    _Float16* Kh = Qh + per;
    _Float16* Vt = Kh + per;

    // 3 mats * 2048 mtiles * 6 ntiles = 36864 waves / 8 per block
    proj_kernel<<<4608, 256, 0, stream>>>(x, Wq, Wk, Wv, Qh, Kh, Vt);
    // 16 batches * 128 qtiles = 2048 waves / 4 per block
    attn_kernel<<<512, 32 * ATT_WAVES, 0, stream>>>(Qh, Kh, Vt, out);
}
